// LpAlignEntropyLoss_58463094833405
// MI455X (gfx1250) — matrix operand order repaired and checked by reading, not compiled, not run on hardware
//
#include <hip/hip_runtime.h>

#define BN 2048
#define DD 128
#define NVIEWS 2
#define EPSV 1e-8f
#define LOG_NM1 7.62413120f   // log(2047)

typedef __attribute__((ext_vector_type(16))) __bf16 v16bf;
typedef __attribute__((ext_vector_type(8)))  __bf16 v8bf;
typedef __attribute__((ext_vector_type(4)))  __bf16 v4bf;
typedef __attribute__((ext_vector_type(8)))  float  v8f;

// ---------------------------------------------------------------------------
// Zero the two scalar accumulators (workspace is poisoned with 0xAA).
// ---------------------------------------------------------------------------
__global__ void init_kernel(float* acc) {
  if (threadIdx.x < 2) acc[threadIdx.x] = 0.f;
}

// ---------------------------------------------------------------------------
// Pass 1: one wave per row (8 rows per 256-thread block).
//  - split fp32 row into bf16 hi + bf16 lo (stored to workspace)
//  - row squared-norm n and row sum s (fp32 exact)
//  - for view 0: alignment term ||z0 - z1 + eps|| accumulated atomically
// ---------------------------------------------------------------------------
__global__ __launch_bounds__(256) void prep_kernel(
    const float* __restrict__ z0, const float* __restrict__ z1,
    float* __restrict__ nrm, float* __restrict__ rsum,
    __bf16* __restrict__ Ahi, __bf16* __restrict__ Alo,
    float* __restrict__ align_acc) {
  const int lane = threadIdx.x & 31;
  const int wv   = threadIdx.x >> 5;
  const int g    = blockIdx.x * 8 + wv;     // 0 .. 2*BN-1
  const int view = g >> 11;
  const int r    = g & (BN - 1);
  const float* __restrict__ z = view ? z1 : z0;

  const float4 x = *(const float4*)(z + (size_t)r * DD + lane * 4);
  float xs[4] = {x.x, x.y, x.z, x.w};

  v4bf h4, l4;
  float n = 0.f, s = 0.f;
#pragma unroll
  for (int c = 0; c < 4; ++c) {
    float xv = xs[c];
    __bf16 h = (__bf16)xv;
    __bf16 l = (__bf16)(xv - (float)h);   // residual for bf16x3 GEMM
    h4[c] = h; l4[c] = l;
    n = fmaf(xv, xv, n);
    s += xv;
  }
  const size_t off = (size_t)g * DD + lane * 4;
  *(v4bf*)(Ahi + off) = h4;
  *(v4bf*)(Alo + off) = l4;

  float a = 0.f;
  if (view == 0) {
    const float4 y = *(const float4*)(z1 + (size_t)r * DD + lane * 4);
    float ys[4] = {y.x, y.y, y.z, y.w};
#pragma unroll
    for (int c = 0; c < 4; ++c) {
      float d = xs[c] - ys[c] + EPSV;     // eps added elementwise, per reference
      a = fmaf(d, d, a);
    }
  }
#pragma unroll
  for (int m = 1; m < 32; m <<= 1) {
    n += __shfl_xor(n, m, 32);
    s += __shfl_xor(s, m, 32);
    a += __shfl_xor(a, m, 32);
  }
  if (lane == 0) {
    nrm[g]  = n;
    rsum[g] = s;
    if (view == 0) atomicAdd(align_acc, __builtin_amdgcn_sqrtf(a));
  }
}

// ---------------------------------------------------------------------------
// Pass 2: one block per (view, 16-row i-tile). 4 waves split the j dimension.
// G = Z Z^T via bf16 hi/lo split; THREE independent WMMA accumulators
// (hi*hi, hi*lo, lo*hi) so consecutive v_wmma ops never RAW-chain within a
// kt group. Fused dist -> exp -> row-sum epilogue; logsumexp per row.
// ---------------------------------------------------------------------------
__global__ __launch_bounds__(128) void gram_lse_kernel(
    const float* __restrict__ nrm, const float* __restrict__ rsum,
    const __bf16* __restrict__ Ahi, const __bf16* __restrict__ Alo,
    float* __restrict__ ent_acc) {
  const int lane = threadIdx.x & 31;
  const int wv   = threadIdx.x >> 5;   // 0..3
  const int half = lane >> 4;          // 0 or 1
  const int l16  = lane & 15;
  const int view = blockIdx.x >> 7;
  const int i0   = (blockIdx.x & 127) << 4;

  const __bf16* __restrict__ ZH = Ahi + (size_t)view * BN * DD;
  const __bf16* __restrict__ ZL = Alo + (size_t)view * BN * DD;
  const float*  __restrict__ nv = nrm  + view * BN;
  const float*  __restrict__ sv = rsum + view * BN;

  // ---- A fragments (rows i0..i0+15, full K=128), per-lane layout:
  // lane<16: K {0..15} in elems 0..15 (per 32-chunk);
  // lane>=16: K {16..31}.
  v16bf ah[4], al[4];
  {
    const __bf16* rh = ZH + (size_t)(i0 + l16) * DD + half * 16;
    const __bf16* rl = ZL + (size_t)(i0 + l16) * DD + half * 16;
#pragma unroll
    for (int kt = 0; kt < 4; ++kt) {
      union { v16bf v; v8bf h[2]; } uh, ul;
      uh.h[0] = *(const v8bf*)(rh + kt * 32);
      uh.h[1] = *(const v8bf*)(rh + kt * 32 + 8);
      ul.h[0] = *(const v8bf*)(rl + kt * 32);
      ul.h[1] = *(const v8bf*)(rl + kt * 32 + 8);
      ah[kt] = uh.v; al[kt] = ul.v;
    }
  }
  float ni[8], si[8];
#pragma unroll
  for (int v = 0; v < 8; ++v) {
    const int m = i0 + half * 8 + v;
    ni[v] = nv[m];
    si[v] = sv[m];
  }

  float rowe[8];
#pragma unroll
  for (int v = 0; v < 8; ++v) rowe[v] = 0.f;

  for (int j0 = wv * 16; j0 < BN; j0 += 64) {
    const float nj = nv[j0 + l16];
    const float sj = sv[j0 + l16];
    // B fragment: lane holds col N=l16 -> row (j0+l16); K contiguous 16 elems
    // starting at kt*32 + half*16 (same k order as the A fragments).
    const __bf16* bhp = ZH + (size_t)(j0 + l16) * DD + half * 16;
    const __bf16* blp = ZL + (size_t)(j0 + l16) * DD + half * 16;

    v8f acc0 = {};   // hi * hi
    v8f acc1 = {};   // hi * lo
    v8f acc2 = {};   // lo * hi
#pragma unroll
    for (int kt = 0; kt < 4; ++kt) {
      v16bf Bh = *(const v16bf*)(bhp + kt * 32);
      v16bf Bl = *(const v16bf*)(blp + kt * 32);
      acc0 = __builtin_amdgcn_wmma_f32_16x16x32_bf16(
          false, ah[kt], false, Bh, (short)0, acc0, false, false);
      acc1 = __builtin_amdgcn_wmma_f32_16x16x32_bf16(
          false, ah[kt], false, Bl, (short)0, acc1, false, false);
      acc2 = __builtin_amdgcn_wmma_f32_16x16x32_bf16(
          false, al[kt], false, Bh, (short)0, acc2, false, false);
    }
    // Epilogue: dist^2 = n_i + n_j - 2G + 2eps(s_i - s_j) + D*eps^2
#pragma unroll
    for (int v = 0; v < 8; ++v) {
      const float g2 = acc0[v] + acc1[v] + acc2[v];
      float d2 = ni[v] + nj - 2.f * g2
               + 2.f * EPSV * (si[v] - sj) + (float)DD * EPSV * EPSV;
      d2 = fmaxf(d2, 0.f);
      float e = __expf(-__builtin_amdgcn_sqrtf(d2));   // tau = 1
      if (i0 + half * 8 + v == j0 + l16) e = 0.f;      // diagonal mask
      rowe[v] += e;
    }
  }

  // Reduce the 16 lanes of each half (rows stay within their half for xor<16)
#pragma unroll
  for (int v = 0; v < 8; ++v) {
#pragma unroll
    for (int m = 1; m < 16; m <<= 1)
      rowe[v] += __shfl_xor(rowe[v], m, 32);
  }

  __shared__ float red[4][16];
  if (l16 == 0) {
#pragma unroll
    for (int v = 0; v < 8; ++v) red[wv][half * 8 + v] = rowe[v];
  }
  __syncthreads();

  if (threadIdx.x < 16) {
    const float t = red[0][threadIdx.x] + red[1][threadIdx.x]
                  + red[2][threadIdx.x] + red[3][threadIdx.x];
    float val = __logf(t) - LOG_NM1;          // log-mean-exp for one row
#pragma unroll
    for (int m = 1; m < 16; m <<= 1)
      val += __shfl_xor(val, m, 32);
    if (threadIdx.x == 0) atomicAdd(ent_acc, val);
  }
}

// ---------------------------------------------------------------------------
// out = align/B - entropy/(2B)
// ---------------------------------------------------------------------------
__global__ void finalize_kernel(const float* acc, float* out) {
  if (threadIdx.x == 0)
    out[0] = acc[0] / (float)BN - acc[1] / (float)(NVIEWS * BN);
}

extern "C" void kernel_launch(void* const* d_in, const int* in_sizes, int n_in,
                              void* d_out, int out_size, void* d_ws, size_t ws_size,
                              hipStream_t stream) {
  (void)in_sizes; (void)n_in; (void)out_size; (void)ws_size;
  const float* z0 = (const float*)d_in[0];
  const float* z1 = (const float*)d_in[1];
  float* out = (float*)d_out;

  // workspace layout (all offsets keep 32B alignment for v16bf loads)
  float*  acc  = (float*)d_ws;                      // [0]=align, [1]=entropy
  float*  nrm  = acc + 8;                           // 2*BN
  float*  rsum = nrm + NVIEWS * BN;                 // 2*BN
  __bf16* Ahi  = (__bf16*)(rsum + NVIEWS * BN);     // 2*BN*DD bf16
  __bf16* Alo  = Ahi + (size_t)NVIEWS * BN * DD;    // 2*BN*DD bf16

  init_kernel<<<1, 32, 0, stream>>>(acc);
  prep_kernel<<<(NVIEWS * BN) / 8, 256, 0, stream>>>(z0, z1, nrm, rsum, Ahi, Alo, acc);
  gram_lse_kernel<<<NVIEWS * (BN / 16), 128, 0, stream>>>(nrm, rsum, Ahi, Alo, acc + 1);
  finalize_kernel<<<1, 32, 0, stream>>>(acc, out);
}
